// Former2Conv_30030411334031
// MI455X (gfx1250) — matrix operand order repaired and checked by reading, not compiled, not run on hardware
//
#include <hip/hip_runtime.h>

// ---------------------------------------------------------------------------
// Former2Conv cross-attention, MI455X (gfx1250, wave32, WMMA f16->f32).
// Shapes: b=8, heads=8, c=64, hw=4096, m=256, DIM=256, INNER=512.
// Key simplification: the torch raw-view reshapes mean only 512 distinct
// Q rows per (b,head) (groups of 8 output rows are identical pre-residual),
// and K/V per (b,head) are contiguous chunks of z@W. Total ~3.5 GFLOP,
// ~20MB traffic -> matrix-core bound; every GEMM uses v_wmma_f32_16x16x32_f16.
// This revision software-pipelines all fragment loads (1-deep prefetch) so
// WMMAs overlap global/LDS load latency instead of s_wait_loadcnt 0x0 stalls.
// ---------------------------------------------------------------------------

typedef __attribute__((ext_vector_type(16))) _Float16 v16h;
typedef __attribute__((ext_vector_type(8)))  float    v8f;

#if defined(__has_builtin) && __has_builtin(__builtin_amdgcn_exp2f)
#define EXP2F __builtin_amdgcn_exp2f
#else
#define EXP2F exp2f
#endif

union FragH {
    v16h   h;
    float4 f4[2];
};

__device__ __forceinline__ v8f wmma16(v16h a, v16h b, v8f c) {
    // D = A(16x32 f16) * B(32x16 f16) + C(16x16 f32)
    return __builtin_amdgcn_wmma_f32_16x16x32_f16(
        /*neg_a=*/false, a, /*neg_b=*/false, b,
        /*c_mod=*/(short)0, c, /*reuse_a=*/false, /*reuse_b=*/false);
}

// A-fragment from f16 row-major [row][k], ld in halves.
// Lane L: M = row0 + L%16; halves 0..7 = K[g*8 .. g*8+7], 8..15 = K[16+g*8 ..]
__device__ __forceinline__ v16h load_a_h(const _Float16* p, int ld, int row0, int k0) {
    int lane = threadIdx.x & 31;
    const _Float16* r = p + (row0 + (lane & 15)) * ld + k0 + (lane >> 4) * 8;
    FragH f;
    f.f4[0] = *(const float4*)(r);
    f.f4[1] = *(const float4*)(r + 16);
    return f.h;
}

// A-fragment from f32 row-major source, converted to f16 on the fly.
__device__ __forceinline__ v16h load_a_f(const float* p, int ld, int row0, int k0) {
    int lane = threadIdx.x & 31;
    const float* r = p + (row0 + (lane & 15)) * ld + k0 + (lane >> 4) * 8;
    float4 a0 = *(const float4*)(r);
    float4 a1 = *(const float4*)(r + 4);
    float4 b0 = *(const float4*)(r + 16);
    float4 b1 = *(const float4*)(r + 20);
    v16h v;
    v[0]  = (_Float16)a0.x; v[1]  = (_Float16)a0.y; v[2]  = (_Float16)a0.z; v[3]  = (_Float16)a0.w;
    v[4]  = (_Float16)a1.x; v[5]  = (_Float16)a1.y; v[6]  = (_Float16)a1.z; v[7]  = (_Float16)a1.w;
    v[8]  = (_Float16)b0.x; v[9]  = (_Float16)b0.y; v[10] = (_Float16)b0.z; v[11] = (_Float16)b0.w;
    v[12] = (_Float16)b1.x; v[13] = (_Float16)b1.y; v[14] = (_Float16)b1.z; v[15] = (_Float16)b1.w;
    return v;
}

// B-fragment from BT stored as [n][k] row-major (k contiguous), ld in halves.
// Lane L: N = n0 + L%16; halves 0..7 = K[g*8 .. g*8+7], 8..15 = K[16+g*8 ..]
__device__ __forceinline__ v16h load_bt(const _Float16* p, int ld, int n0, int k0) {
    int lane = threadIdx.x & 31;
    const _Float16* r = p + (n0 + (lane & 15)) * ld + k0 + (lane >> 4) * 8;
    FragH f;
    f.f4[0] = *(const float4*)(r);
    f.f4[1] = *(const float4*)(r + 16);
    return f.h;
}

// ---------------------------------------------------------------------------
// Kernel 0: weight transposes to f16.  WkT/WvT: [512][256], WoT: [64][512].
// ---------------------------------------------------------------------------
__global__ void __launch_bounds__(256) prep_kernel(
    const float* __restrict__ Wk, const float* __restrict__ Wv,
    const float* __restrict__ Wo,
    _Float16* __restrict__ WkT, _Float16* __restrict__ WvT,
    _Float16* __restrict__ WoT) {
    int idx = blockIdx.x * 256 + threadIdx.x;
    if (idx < 131072) {                 // 512 x 256
        int n = idx >> 8, d = idx & 255;
        WkT[idx] = (_Float16)Wk[d * 512 + n];
        WvT[idx] = (_Float16)Wv[d * 512 + n];
    } else {
        int t = idx - 131072;           // 64 x 512
        int cc = t >> 9, k = t & 511;
        WoT[t] = (_Float16)Wo[k * 64 + cc];
    }
}

// ---------------------------------------------------------------------------
// Kernel 1: klin = z@Wk+bk, vlin = z@Wv+bv per batch (256x256 @ 256x512).
// Stores K row-major f16 (which IS the per-head 256x64 layout) and V
// transposed per head: Vt[(b*8+hh)*64 + cc][mm].
// Grid: 8b x 16 rowtiles x 2 colgroups = 256 blocks, 4 waves/block.
// Double-buffered fragments: loads for k-step ks+1 issue before the 8 WMMAs
// of k-step ks consume their operands.
// ---------------------------------------------------------------------------
__global__ void __launch_bounds__(128) kv_kernel(
    const float* __restrict__ z, const float* __restrict__ bk,
    const float* __restrict__ bv,
    const _Float16* __restrict__ WkT, const _Float16* __restrict__ WvT,
    _Float16* __restrict__ Kh, _Float16* __restrict__ Vt) {
    int blk = blockIdx.x;
    int b = blk >> 5;
    int rem = blk & 31;
    int row0 = (rem >> 1) * 16;
    int wave = threadIdx.x >> 5, lane = threadIdx.x & 31;
    int n0 = (rem & 1) * 256 + wave * 64;
    const float* zb = z + b * 65536;

    v8f ak[4] = {};
    v8f av[4] = {};

    v16h a_cur = load_a_f(zb, 256, row0, 0);
    v16h bk_cur[4], bv_cur[4];
#pragma unroll
    for (int t = 0; t < 4; ++t) {
        bk_cur[t] = load_bt(WkT, 256, n0 + t * 16, 0);
        bv_cur[t] = load_bt(WvT, 256, n0 + t * 16, 0);
    }
#pragma unroll
    for (int ks = 0; ks < 8; ++ks) {
        v16h a_nxt;
        v16h bk_nxt[4], bv_nxt[4];
        if (ks < 7) {
            int k1 = (ks + 1) * 32;
            a_nxt = load_a_f(zb, 256, row0, k1);
#pragma unroll
            for (int t = 0; t < 4; ++t) {
                bk_nxt[t] = load_bt(WkT, 256, n0 + t * 16, k1);
                bv_nxt[t] = load_bt(WvT, 256, n0 + t * 16, k1);
            }
        }
#pragma unroll
        for (int t = 0; t < 4; ++t) {
            ak[t] = wmma16(a_cur, bk_cur[t], ak[t]);
            av[t] = wmma16(a_cur, bv_cur[t], av[t]);
        }
        if (ks < 7) {
            a_cur = a_nxt;
#pragma unroll
            for (int t = 0; t < 4; ++t) {
                bk_cur[t] = bk_nxt[t];
                bv_cur[t] = bv_nxt[t];
            }
        }
    }

    int g = lane >> 4, nl = lane & 15;
#pragma unroll
    for (int t = 0; t < 4; ++t) {
        int col = n0 + t * 16 + nl;
        float bkv = bk[col], bvv = bv[col];
#pragma unroll
        for (int v = 0; v < 8; ++v) {
            int r = row0 + g * 8 + v;                      // klin row in [0,256)
            Kh[(b * 256 + r) * 512 + col] = (_Float16)(ak[t][v] + bkv);
            int hh = r >> 5;
            int mm = (r & 31) * 8 + (col >> 6);
            int cc = col & 63;
            Vt[((b * 8 + hh) * 64 + cc) * 256 + mm] = (_Float16)(av[t][v] + bvv);
        }
    }
}

// ---------------------------------------------------------------------------
// Kernel 2: attention over the 512 distinct Q rows per (b,head).
// Grid: 8b x 8hh x 8 rowgroups = 512 blocks; 4 waves, wave = 16 Q rows.
// S = Q*K^T (16 tiles of 16x16), register softmax (exp2 + shfl_xor over the
// 16-lane group), P -> LDS f16, O = P*V, normalize, store f16 in the layout
// the projection GEMM wants: O[(b*512+r)*512 + hh*64 + cc].
// Both GEMM loops are flattened with 1-deep fragment prefetch.
// ---------------------------------------------------------------------------
__global__ void __launch_bounds__(128) attn_kernel(
    const float* __restrict__ x, const _Float16* __restrict__ Kh,
    const _Float16* __restrict__ Vt, _Float16* __restrict__ O) {
    __shared__ _Float16 Plds[4][16][256];

    int blk = blockIdx.x;
    int b = blk >> 6;
    int hh = (blk >> 3) & 7;
    int rg = blk & 7;
    int wave = threadIdx.x >> 5, lane = threadIdx.x & 31;
    int qr0 = rg * 64 + wave * 16;

    const float*    Q = x + b * 262144 + hh * 32768;        // 512 x 64 f32
    const _Float16* K = Kh + (b * 8 + hh) * 16384;          // 256 x 64 f16
    const _Float16* V = Vt + (b * 8 + hh) * 16384;          // 64 x 256 f16 (V^T)

    // ---- S = Q * K^T, 16 rows x 256 cols in 16 accumulator tiles ----
    // i = ks*16 + n over (ks in 0..1, n in 0..15), 1-deep prefetch.
    v8f s[16] = {};
    {
        v16h aS = load_a_f(Q, 64, qr0, 0);
        v16h bS = load_bt(K, 64, 0, 0);
#pragma unroll
        for (int i = 0; i < 32; ++i) {
            int n = i & 15;
            v16h bN, aN;
            if (i < 31)
                bN = load_bt(K, 64, ((i + 1) & 15) * 16, ((i + 1) >> 4) * 32);
            if (i == 15)
                aN = load_a_f(Q, 64, qr0, 32);
            s[n] = wmma16(aS, bS, s[n]);
            if (i < 31) bS = bN;
            if (i == 15) aS = aN;
        }
    }

    // ---- softmax (scale folded into log2 domain) ----
    const float SC = 0.125f * 1.44269504088896340736f;      // c^-0.5 * log2(e)
    int g = lane >> 4, nl = lane & 15;
    float rinv[8];
#pragma unroll
    for (int v = 0; v < 8; ++v) {
        float m = -3.0e38f;
#pragma unroll
        for (int n = 0; n < 16; ++n) m = fmaxf(m, s[n][v]);
        m *= SC;
#pragma unroll
        for (int off = 1; off < 16; off <<= 1)
            m = fmaxf(m, __shfl_xor(m, off, 16));
        float sum = 0.f;
#pragma unroll
        for (int n = 0; n < 16; ++n) {
            float p = EXP2F(s[n][v] * SC - m);
            s[n][v] = p;
            sum += p;
        }
#pragma unroll
        for (int off = 1; off < 16; off <<= 1)
            sum += __shfl_xor(sum, off, 16);
        rinv[v] = 1.0f / sum;
    }

    // ---- P -> LDS f16 (layout change: D-frag -> A-frag) ----
#pragma unroll
    for (int n = 0; n < 16; ++n)
#pragma unroll
        for (int v = 0; v < 8; ++v)
            Plds[wave][g * 8 + v][n * 16 + nl] = (_Float16)s[n][v];
    __syncthreads();

    // ---- O = P * V ----
    // i = ks*4 + t over (ks in 0..7, t in 0..3), 1-deep prefetch.
    v8f o[4] = {};
    {
        const _Float16* Pw = &Plds[wave][0][0];
        v16h aP = load_a_h(Pw, 256, 0, 0);
        v16h bP = load_bt(V, 256, 0, 0);
#pragma unroll
        for (int i = 0; i < 32; ++i) {
            int t = i & 3;
            v16h bN, aN;
            if (i < 31)
                bN = load_bt(V, 256, ((i + 1) & 3) * 16, ((i + 1) >> 2) * 32);
            if (t == 3 && i < 31)
                aN = load_a_h(Pw, 256, 0, ((i >> 2) + 1) * 32);
            o[t] = wmma16(aP, bP, o[t]);
            if (i < 31) bP = bN;
            if (t == 3 && i < 31) aP = aN;
        }
    }

#pragma unroll
    for (int t = 0; t < 4; ++t) {
        int cc = t * 16 + nl;
#pragma unroll
        for (int v = 0; v < 8; ++v) {
            int r = qr0 + g * 8 + v;                        // distinct row in [0,512)
            O[(b * 512 + r) * 512 + hh * 64 + cc] = (_Float16)(o[t][v] * rinv[v]);
        }
    }
}

// ---------------------------------------------------------------------------
// Kernel 3: projection (512x512 f16) @ Wo (512x64) + bo -> f32 distinct rows.
// Grid: 8b x 8 rowgroups = 64 blocks; 4 waves, wave = 16 rows x 64 cols.
// Flattened i = ks*4 + t over (ks in 0..15, t in 0..3), 1-deep prefetch.
// ---------------------------------------------------------------------------
__global__ void __launch_bounds__(128) proj_kernel(
    const _Float16* __restrict__ O, const _Float16* __restrict__ WoT,
    const float* __restrict__ bo, float* __restrict__ PD) {
    int blk = blockIdx.x;
    int b = blk >> 3, rg = blk & 7;
    int wave = threadIdx.x >> 5, lane = threadIdx.x & 31;
    int pr0 = rg * 64 + wave * 16;
    const _Float16* A = O + b * 262144;                     // 512 x 512 f16

    v8f acc[4] = {};
    {
        v16h aP = load_a_h(A, 512, pr0, 0);
        v16h bP = load_bt(WoT, 512, 0, 0);
#pragma unroll
        for (int i = 0; i < 64; ++i) {
            int t = i & 3;
            v16h bN, aN;
            if (i < 63)
                bN = load_bt(WoT, 512, ((i + 1) & 3) * 16, ((i + 1) >> 2) * 32);
            if (t == 3 && i < 63)
                aN = load_a_h(A, 512, pr0, ((i >> 2) + 1) * 32);
            acc[t] = wmma16(aP, bP, acc[t]);
            if (i < 63) bP = bN;
            if (t == 3 && i < 63) aP = aN;
        }
    }

    int g = lane >> 4, nl = lane & 15;
#pragma unroll
    for (int t = 0; t < 4; ++t) {
        int cc = t * 16 + nl;
        float bb = bo[cc];
#pragma unroll
        for (int v = 0; v < 8; ++v) {
            int r = pr0 + g * 8 + v;
            PD[(b * 512 + r) * 64 + cc] = acc[t][v] + bb;
        }
    }
}

// ---------------------------------------------------------------------------
// Kernel 4: broadcast each distinct row to its 8 output rows + residual.
// Fully coalesced float4 path for the mandatory 16MB of traffic.
// ---------------------------------------------------------------------------
__global__ void __launch_bounds__(256) expand_kernel(
    const float* __restrict__ x, const float* __restrict__ PD,
    float* __restrict__ y) {
    int i = blockIdx.x * 256 + threadIdx.x;                 // float4 index
    int f = i << 2;
    int cc = f & 63;
    int l = (f >> 6) & 4095;
    int b = f >> 18;
    float4 xv = ((const float4*)x)[i];
    float4 pv = *(const float4*)(PD + (b * 512 + (l >> 3)) * 64 + cc);
    float4 r;
    r.x = xv.x + pv.x; r.y = xv.y + pv.y;
    r.z = xv.z + pv.z; r.w = xv.w + pv.w;
    ((float4*)y)[i] = r;
}

// ---------------------------------------------------------------------------
extern "C" void kernel_launch(void* const* d_in, const int* in_sizes, int n_in,
                              void* d_out, int out_size, void* d_ws, size_t ws_size,
                              hipStream_t stream) {
    (void)in_sizes; (void)n_in; (void)out_size; (void)ws_size;
    const float* x  = (const float*)d_in[0];
    const float* z  = (const float*)d_in[1];
    const float* Wk = (const float*)d_in[2];
    const float* bk = (const float*)d_in[3];
    const float* Wv = (const float*)d_in[4];
    const float* bv = (const float*)d_in[5];
    const float* Wo = (const float*)d_in[6];
    const float* bo = (const float*)d_in[7];
    float* out = (float*)d_out;

    char* ws = (char*)d_ws;                                 // ~9.6 MB used
    _Float16* WkT = (_Float16*)(ws);                        //  256 KB
    _Float16* WvT = (_Float16*)(ws + 262144);               //  256 KB
    _Float16* WoT = (_Float16*)(ws + 524288);               //   64 KB
    _Float16* Kh  = (_Float16*)(ws + 589824);               //    2 MB
    _Float16* Vt  = (_Float16*)(ws + 2686976);              //    2 MB
    _Float16* O   = (_Float16*)(ws + 4784128);              //    4 MB
    float*    PD  = (float*)   (ws + 8978432);              //    1 MB

    prep_kernel  <<< 640, 256, 0, stream>>>(Wk, Wv, Wo, WkT, WvT, WoT);
    kv_kernel    <<< 256, 128, 0, stream>>>(z, bk, bv, WkT, WvT, Kh, Vt);
    attn_kernel  <<< 512, 128, 0, stream>>>(x, Kh, Vt, O);
    proj_kernel  <<<  64, 128, 0, stream>>>(O, WoT, bo, PD);
    expand_kernel<<<2048, 256, 0, stream>>>(x, PD, out);
}
